// Decoder8_13348758356164
// MI455X (gfx1250) — matrix operand order repaired and checked by reading, not compiled, not run on hardware
//
#include <hip/hip_runtime.h>
#include <math.h>

// ---------------------------------------------------------------------------
// Dims from the reference
// ---------------------------------------------------------------------------
#define BSZ    32
#define SEQ    512
#define HID    768
#define INDIM  828      // 768 + 30 + 30
#define HIDDEN 300
#define ATTN   64
#define FF     600
#define LA     256
#define LBL    200
#define NLAB   42
#define TOK    (BSZ*SEQ)   // 16384
#define POSD   30
#define NEGV   (-1e9f)

typedef __attribute__((ext_vector_type(16))) _Float16 v16h;
typedef __attribute__((ext_vector_type(8)))  _Float16 v8h;
typedef __attribute__((ext_vector_type(8)))  float    v8f;

__device__ __forceinline__ float gelu_f(float x) {
    return 0.5f * x * (1.0f + erff(x * 0.70710678118654752f));
}

// ---------------------------------------------------------------------------
// Generic batched GEMM via v_wmma_f32_16x16x32_f16.
//   C[b] = epilogue( scale * (A[b] @ B[b] + bias) )  [+ residual]
//
// block = 256 threads = 8 wave32s; block tile 64x64. Each wave owns a 16x32
// output strip: ONE A fragment feeds TWO WMMAs (A reuse, 2 matrix ops per
// barrier pair). A staged in LDS as [m][k] f16, B staged TRANSPOSED as
// Bt[n][k] so every fragment read is a contiguous 16B ds_load_b128.
// Row pitch = 40 halves (80B) -> bank offset 20*row mod 64, conflict-free.
// Interior tiles take an unguarded load path; a one-stage register prefetch
// overlaps the next slab's VMEM with the WMMAs. TRANSB is a template arg so
// each instantiation carries only its own staging path.
// act: 0 = none, 1 = GELU (applied before residual add, matching reference).
// ---------------------------------------------------------------------------
template <int TRANSB>
__global__ void gemm_wmma(const float* __restrict__ A, long sA, int lda,
                          const float* __restrict__ B, long sB, int ldb,
                          const float* __restrict__ bias,
                          const float* __restrict__ resid, long sR, int ldr,
                          float* __restrict__ C, long sC, int ldc,
                          int M, int N, int K, float scale, int act) {
    A += (long)blockIdx.z * sA;
    B += (long)blockIdx.z * sB;
    C += (long)blockIdx.z * sC;
    if (resid) resid += (long)blockIdx.z * sR;

    __shared__ alignas(16) _Float16 As[64][40];   // [m][k], pitch 40 halves
    __shared__ alignas(16) _Float16 Bt[64][40];   // [n][k], pitch 40 halves

    const int m0 = blockIdx.y * 64, n0 = blockIdx.x * 64;
    const int t = threadIdx.x;           // 0..255
    const int wave = t >> 5, lane = t & 31;
    const int wm = (wave >> 1) * 16;     // wave's row offset   (0,16,32,48)
    const int wn = (wave & 1) * 32;      // wave's col offset   (0,32)
    const int half = lane >> 4;          // lane group 0/1
    const int l16  = lane & 15;

    // staging maps (8 elements per thread per matrix)
    const int ar = t >> 2, ak = (t & 3) * 8;       // A: row ar, k ak..ak+7
    const int br = t >> 2, bk = (t & 3) * 8;       // B (TRANSB): n row, k run
    const int ck = t >> 3, cn = (t & 7) * 8;       // B (!TRANSB): k row, n run
    const bool mFull = (m0 + 64 <= M);
    const bool nFull = (n0 + 64 <= N);

    const int nk = (K + 31) >> 5;
    float fa[8], fb[8];

    auto loadA = [&](int k0, float* fr) {
        if (mFull && (k0 + 32 <= K)) {
            const float* p = A + (long)(m0 + ar) * lda + k0 + ak;
            #pragma unroll
            for (int e = 0; e < 8; ++e) fr[e] = p[e];
        } else {
            int gm = m0 + ar;
            #pragma unroll
            for (int e = 0; e < 8; ++e) {
                int gk = k0 + ak + e;
                fr[e] = (gm < M && gk < K) ? A[(long)gm * lda + gk] : 0.0f;
            }
        }
    };
    auto loadB = [&](int k0, float* fr) {
        if constexpr (TRANSB) {     // Bsrc[n][k]: 8 contiguous k per thread
            if (nFull && (k0 + 32 <= K)) {
                const float* p = B + (long)(n0 + br) * ldb + k0 + bk;
                #pragma unroll
                for (int e = 0; e < 8; ++e) fr[e] = p[e];
            } else {
                int gn = n0 + br;
                #pragma unroll
                for (int e = 0; e < 8; ++e) {
                    int gk = k0 + bk + e;
                    fr[e] = (gn < N && gk < K) ? B[(long)gn * ldb + gk] : 0.0f;
                }
            }
        } else {                    // Bsrc[k][n]: 8 contiguous n per thread
            if (nFull && (k0 + 32 <= K)) {
                const float* p = B + (long)(k0 + ck) * ldb + n0 + cn;
                #pragma unroll
                for (int e = 0; e < 8; ++e) fr[e] = p[e];
            } else {
                int gk = k0 + ck;
                #pragma unroll
                for (int e = 0; e < 8; ++e) {
                    int gn = n0 + cn + e;
                    fr[e] = (gk < K && gn < N) ? B[(long)gk * ldb + gn] : 0.0f;
                }
            }
        }
    };

    v8f acc0 = {}, acc1 = {};
    loadA(0, fa);
    loadB(0, fb);

    for (int it = 0; it < nk; ++it) {
        // ---- registers -> LDS (f32 -> f16)
        {
            v8h ha;
            #pragma unroll
            for (int e = 0; e < 8; ++e) ha[e] = (_Float16)fa[e];
            *(v8h*)&As[ar][ak] = ha;                 // ds_store_b128
            if constexpr (TRANSB) {
                v8h hb;
                #pragma unroll
                for (int e = 0; e < 8; ++e) hb[e] = (_Float16)fb[e];
                *(v8h*)&Bt[br][bk] = hb;             // ds_store_b128
            } else {
                #pragma unroll
                for (int e = 0; e < 8; ++e)          // scatter to Bt[n][k]
                    Bt[cn + e][ck] = (_Float16)fb[e];
            }
        }
        __syncthreads();

        // ---- prefetch next slab while the WMMAs + LDS reads run
        float fan[8], fbn[8];
        if (it + 1 < nk) {
            loadA((it + 1) * 32, fan);
            loadB((it + 1) * 32, fbn);
        }

        // ---- fragments (contiguous k = 16*half + s for A and B), 16B LDS loads each
        const int mr = wm + l16, nc0 = wn + l16, nc1 = wn + 16 + l16;
        v8h alo = *(const v8h*)&As[mr][16 * half];
        v8h ahi = *(const v8h*)&As[mr][16 * half + 8];
        v16h a = __builtin_shufflevector(alo, ahi, 0,1,2,3,4,5,6,7,8,9,10,11,12,13,14,15);
        v8h b0lo = *(const v8h*)&Bt[nc0][16 * half];
        v8h b0hi = *(const v8h*)&Bt[nc0][16 * half + 8];
        v16h b0 = __builtin_shufflevector(b0lo, b0hi, 0,1,2,3,4,5,6,7,8,9,10,11,12,13,14,15);
        v8h b1lo = *(const v8h*)&Bt[nc1][16 * half];
        v8h b1hi = *(const v8h*)&Bt[nc1][16 * half + 8];
        v16h b1 = __builtin_shufflevector(b1lo, b1hi, 0,1,2,3,4,5,6,7,8,9,10,11,12,13,14,15);
        acc0 = __builtin_amdgcn_wmma_f32_16x16x32_f16(
            false, a, false, b0, (short)0, acc0, false, false);
        acc1 = __builtin_amdgcn_wmma_f32_16x16x32_f16(
            false, a, false, b1, (short)0, acc1, false, false);
        __syncthreads();

        if (it + 1 < nk) {
            #pragma unroll
            for (int e = 0; e < 8; ++e) { fa[e] = fan[e]; fb[e] = fbn[e]; }
        }
    }

    // ---- epilogue + store (C/D layout: vgpr r -> M = r + 8*half, N = lane&15)
    const int gn0 = n0 + wn + l16, gn1 = gn0 + 16;
    #pragma unroll
    for (int r = 0; r < 8; ++r) {
        int gm = m0 + wm + r + 8 * half;
        if (gm < M) {
            if (gn0 < N) {
                float v = acc0[r];
                if (bias) v += bias[gn0];
                v *= scale;
                if (act == 1) v = gelu_f(v);
                if (resid) v += resid[(long)gm * ldr + gn0];
                C[(long)gm * ldc + gn0] = v;
            }
            if (gn1 < N) {
                float v = acc1[r];
                if (bias) v += bias[gn1];
                v *= scale;
                if (act == 1) v = gelu_f(v);
                if (resid) v += resid[(long)gm * ldr + gn1];
                C[(long)gm * ldc + gn1] = v;
            }
        }
    }
}

// ---------------------------------------------------------------------------
// h = concat(inputs, pos_emb[pos_ids], dist_emb[dist])   [TOK, 828]
// ---------------------------------------------------------------------------
__global__ void build_h(const float* __restrict__ inputs,
                        const int* __restrict__ pos_ids,
                        const int* __restrict__ dist,
                        const float* __restrict__ pos_emb,
                        const float* __restrict__ dist_emb,
                        float* __restrict__ h) {
    int tok = blockIdx.x;
    int pid = pos_ids[tok], did = dist[tok];
    for (int i = threadIdx.x; i < INDIM; i += blockDim.x) {
        float v;
        if (i < HID)             v = inputs[(long)tok * HID + i];
        else if (i < HID + POSD) v = pos_emb[pid * POSD + (i - HID)];
        else                     v = dist_emb[did * POSD + (i - HID - POSD)];
        h[(long)tok * INDIM + i] = v;
    }
}

// ---------------------------------------------------------------------------
// Entity masked max-pool over seq (pos == 300 marks entity tokens)
// ---------------------------------------------------------------------------
__global__ void pool_entities(const float* __restrict__ inputs,
                              const int* __restrict__ subj_pos,
                              const int* __restrict__ obj_pos,
                              float* __restrict__ subj_out,
                              float* __restrict__ obj_out) {
    int b = blockIdx.x;
    for (int d = threadIdx.x; d < HID; d += blockDim.x) {
        float m1 = -1e30f, m2 = -1e30f;
        for (int s = 0; s < SEQ; ++s) {
            float v = inputs[((long)b * SEQ + s) * HID + d];
            if (subj_pos[b * SEQ + s] == 300) m1 = fmaxf(m1, v);
            if (obj_pos[b * SEQ + s]  == 300) m2 = fmaxf(m2, v);
        }
        subj_out[b * HID + d] = m1;
        obj_out[b * HID + d]  = m2;
    }
}

// ---------------------------------------------------------------------------
// In-place masked softmax over attention rows: S[b,q,:], mask==0 -> NEG
// grid = (SEQ, BSZ), block = 256
// ---------------------------------------------------------------------------
__global__ void attn_softmax(float* __restrict__ S, const int* __restrict__ mask) {
    long base = ((long)blockIdx.y * SEQ + blockIdx.x) * SEQ;
    float* row = S + base;
    const int* m = mask + base;
    __shared__ float red[256];
    int tid = threadIdx.x;
    float mx = -3.0e38f;
    for (int c = tid; c < SEQ; c += 256) {
        float v = (m[c] == 0) ? NEGV : row[c];
        row[c] = v;
        mx = fmaxf(mx, v);
    }
    red[tid] = mx; __syncthreads();
    for (int s = 128; s > 0; s >>= 1) {
        if (tid < s) red[tid] = fmaxf(red[tid], red[tid + s]);
        __syncthreads();
    }
    mx = red[0]; __syncthreads();
    float sum = 0.0f;
    for (int c = tid; c < SEQ; c += 256) {
        float e = __expf(row[c] - mx);
        row[c] = e; sum += e;
    }
    red[tid] = sum; __syncthreads();
    for (int s = 128; s > 0; s >>= 1) {
        if (tid < s) red[tid] += red[tid + s];
        __syncthreads();
    }
    float inv = 1.0f / red[0];
    for (int c = tid; c < SEQ; c += 256) row[c] *= inv;
}

// Generic in-place row softmax (rows of length n), one block per row.
__global__ void softmax_row(float* __restrict__ x, int n) {
    float* row = x + (long)blockIdx.x * n;
    __shared__ float red[256];
    int tid = threadIdx.x;
    float mx = -3.0e38f;
    for (int c = tid; c < n; c += 256) mx = fmaxf(mx, row[c]);
    red[tid] = mx; __syncthreads();
    for (int s = 128; s > 0; s >>= 1) { if (tid < s) red[tid] = fmaxf(red[tid], red[tid + s]); __syncthreads(); }
    mx = red[0]; __syncthreads();
    float sum = 0.0f;
    for (int c = tid; c < n; c += 256) { float e = __expf(row[c] - mx); row[c] = e; sum += e; }
    red[tid] = sum; __syncthreads();
    for (int s = 128; s > 0; s >>= 1) { if (tid < s) red[tid] += red[tid + s]; __syncthreads(); }
    float inv = 1.0f / red[0];
    for (int c = tid; c < n; c += 256) row[c] *= inv;
}

// ---------------------------------------------------------------------------
// kt_sum[d] = sum_{l=1..41} (label_emb[l] @ la_k.w)[d] + 41*la_k.b[d]
// ---------------------------------------------------------------------------
__global__ void kt_sum_kernel(const float* __restrict__ label_emb,
                              const float* __restrict__ wk,
                              const float* __restrict__ bk,
                              float* __restrict__ kt) {
    int d = threadIdx.x;                        // 256
    float acc = 41.0f * bk[d];
    for (int l = 1; l < NLAB; ++l) {
        const float* le = label_emb + l * LBL;
        for (int j = 0; j < LBL; ++j) acc += le[j] * wk[j * LA + d];
    }
    kt[d] = acc;
}

// scores[b,s] = dot(Qt[b,s,:], kt_sum) + (rowsum+colsum(adj)==0 ? NEG : 0)
// grid = (SEQ, BSZ), block = 128
__global__ void la_scores(const float* __restrict__ Qt,
                          const float* __restrict__ kt,
                          const int* __restrict__ adj,
                          float* __restrict__ scores) {
    int s = blockIdx.x, b = blockIdx.y, tid = threadIdx.x;
    __shared__ float red[128];
    __shared__ int   redi[128];
    const int* arow = adj + ((long)b * SEQ + s) * SEQ;
    const int* acol = adj + (long)b * SEQ * SEQ + s;
    int cnt = 0;
    for (int k = tid; k < SEQ; k += 128) cnt += arow[k] + acol[(long)k * SEQ];
    const float* q = Qt + ((long)b * SEQ + s) * LA;
    float dot = 0.0f;
    for (int d = tid; d < LA; d += 128) dot += q[d] * kt[d];
    red[tid] = dot; redi[tid] = cnt; __syncthreads();
    for (int t = 64; t > 0; t >>= 1) {
        if (tid < t) { red[tid] += red[tid + t]; redi[tid] += redi[tid + t]; }
        __syncthreads();
    }
    if (tid == 0) scores[b * SEQ + s] = red[0] + (redi[0] == 0 ? NEGV : 0.0f);
}

// features[b, 600+d] = sum_s attn[b,s] * sf[b,s,d]
__global__ void la_context(const float* __restrict__ attn,
                           const float* __restrict__ sf,
                           float* __restrict__ features) {
    int b = blockIdx.x;
    for (int d = threadIdx.x; d < HIDDEN; d += blockDim.x) {
        float acc = 0.0f;
        for (int s = 0; s < SEQ; ++s)
            acc += attn[b * SEQ + s] * sf[((long)b * SEQ + s) * HIDDEN + d];
        features[b * 900 + 600 + d] = acc;
    }
}

// ---------------------------------------------------------------------------
// Host-side launcher
// ---------------------------------------------------------------------------
static inline void g_launch(hipStream_t st,
                            const float* A, long sA, int lda,
                            const float* B, long sB, int ldb, int transB,
                            const float* bias,
                            const float* resid, long sR, int ldr,
                            float* C, long sC, int ldc,
                            int M, int N, int K, float scale, int act, int batch) {
    dim3 grid((N + 63) / 64, (M + 63) / 64, batch), blk(256);
    if (transB)
        gemm_wmma<1><<<grid, blk, 0, st>>>(A, sA, lda, B, sB, ldb, bias,
                                           resid, sR, ldr, C, sC, ldc, M, N, K, scale, act);
    else
        gemm_wmma<0><<<grid, blk, 0, st>>>(A, sA, lda, B, sB, ldb, bias,
                                           resid, sR, ldr, C, sC, ldc, M, N, K, scale, act);
}

extern "C" void kernel_launch(void* const* d_in, const int* in_sizes, int n_in,
                              void* d_out, int out_size, void* d_ws, size_t ws_size,
                              hipStream_t stream) {
    // Inputs in setup_inputs() insertion order; params flattened recursively in
    // their own dict/list insertion order, each linear layer contributing w,b.
    const float* inputs       = (const float*)d_in[0];
    const int*   pos_ids      = (const int*)d_in[1];
    const int*   subj_pos     = (const int*)d_in[2];
    const int*   obj_pos      = (const int*)d_in[3];
    const int*   dist         = (const int*)d_in[4];
    const int*   adj          = (const int*)d_in[5];
    const int*   whole_adj    = (const int*)d_in[6];
    const int*   ancestor_adj = (const int*)d_in[7];
    // d_in[8] = input_mask (unused by reference math)
    const float* label_emb = (const float*)d_in[9];
    const float* pos_emb   = (const float*)d_in[10];
    const float* dist_emb  = (const float*)d_in[11];
    const float* subj_w = (const float*)d_in[12];
    const float* subj_b = (const float*)d_in[13];
    const float* obj_w  = (const float*)d_in[14];
    const float* obj_b  = (const float*)d_in[15];
    const float* laq_w  = (const float*)d_in[16];
    const float* laq_b  = (const float*)d_in[17];
    const float* lak_w  = (const float*)d_in[18];
    const float* lak_b  = (const float*)d_in[19];
    const float* feat_w = (const float*)d_in[20];
    const float* feat_b = (const float*)d_in[21];
    const float* out_w  = (const float*)d_in[22];
    const float* out_b  = (const float*)d_in[23];

    float* ws = (float*)d_ws;
    // ---- workspace layout (floats), with aliasing of dead regions
    size_t off = 0;
    float* H0  = ws + off; off += (size_t)TOK * INDIM;        // 13.57M
    float* QKV = ws + off; off += (size_t)6 * TOK * ATTN;     //  6.29M
    float* S   = ws + off; off += (size_t)BSZ * SEQ * SEQ;    //  8.39M
    float* CTX = ws + off; off += (size_t)TOK * 128;          //  2.10M
    float* Cb  = ws + off; off += (size_t)TOK * HIDDEN;       //  4.92M
    float* H1  = ws + off; off += (size_t)TOK * HIDDEN;       //  4.92M
    float* ktsum    = ws + off; off += LA;
    float* subj_out = ws + off; off += (size_t)BSZ * HID;
    float* obj_out  = ws + off; off += (size_t)BSZ * HID;
    float* features = ws + off; off += (size_t)BSZ * 900;
    float* scores   = ws + off; off += (size_t)BSZ * SEQ;
    float* featH    = ws + off; off += (size_t)BSZ * HIDDEN;
    float* outF     = ws + off; off += (size_t)BSZ * LBL;
    float* Fbuf = QKV;  // ff1 output (9.83M) overlays dead QKV+S (14.68M)
    float* Qt   = H0;   // label-attn Q (4.19M) overlays dead H0
    (void)in_sizes; (void)n_in; (void)out_size; (void)ws_size; (void)off;

    float* logits    = (float*)d_out;              // [32,42]
    float* structure = (float*)d_out + BSZ * NLAB; // [32,512,300]

    const float scl = 0.125f;  // 1/sqrt(64)
    const long tA = (long)SEQ * ATTN, tS = (long)SEQ * SEQ, tC = (long)SEQ * 128;

    // 0) h = concat(inputs, pos_emb, dist_emb)
    build_h<<<dim3(TOK), 256, 0, stream>>>(inputs, pos_ids, dist, pos_emb, dist_emb, H0);
    // 1) entity pools + subj/obj features into columns [0,300) and [300,600)
    pool_entities<<<dim3(BSZ), 256, 0, stream>>>(inputs, subj_pos, obj_pos, subj_out, obj_out);
    g_launch(stream, subj_out, 0, HID, subj_w, 0, HIDDEN, 0, subj_b, nullptr, 0, 0,
             features + 0,   0, 900, BSZ, HIDDEN, HID, 1.0f, 0, 1);
    g_launch(stream, obj_out,  0, HID, obj_w,  0, HIDDEN, 0, obj_b,  nullptr, 0, 0,
             features + 300, 0, 900, BSZ, HIDDEN, HID, 1.0f, 0, 1);

    // 2) DGA layers
    for (int layer = 0; layer < 2; ++layer) {
        const float* hin = (layer == 0) ? H0 : H1;
        const int din    = (layer == 0) ? INDIM : HIDDEN;
        float* hout      = (layer == 0) ? H1 : structure;
        const int pbase  = 24 + layer * 18;
        const float** P  = (const float**)(d_in + pbase);
        const float *qf_w = P[0],  *qf_b = P[1],  *kf_w = P[2],  *kf_b = P[3];
        const float *vf_w = P[4],  *vf_b = P[5],  *qb_w = P[6],  *qb_b = P[7];
        const float *kb_w = P[8],  *kb_b = P[9],  *vb_w = P[10], *vb_b = P[11];
        const float *o_w  = P[12], *o_b  = P[13], *f1_w = P[14], *f1_b = P[15];
        const float *f2_w = P[16], *f2_b = P[17];

        float *Qf = QKV, *Kf = QKV + (size_t)TOK*ATTN, *Vf = QKV + (size_t)2*TOK*ATTN;
        float *Qb = QKV + (size_t)3*TOK*ATTN, *Kb = QKV + (size_t)4*TOK*ATTN, *Vb = QKV + (size_t)5*TOK*ATTN;

        g_launch(stream, hin,0,din, qf_w,0,ATTN,0, qf_b, nullptr,0,0, Qf,0,ATTN, TOK,ATTN,din, 1.0f,0,1);
        g_launch(stream, hin,0,din, kf_w,0,ATTN,0, kf_b, nullptr,0,0, Kf,0,ATTN, TOK,ATTN,din, 1.0f,0,1);
        g_launch(stream, hin,0,din, vf_w,0,ATTN,0, vf_b, nullptr,0,0, Vf,0,ATTN, TOK,ATTN,din, 1.0f,0,1);
        g_launch(stream, hin,0,din, qb_w,0,ATTN,0, qb_b, nullptr,0,0, Qb,0,ATTN, TOK,ATTN,din, 1.0f,0,1);
        g_launch(stream, hin,0,din, kb_w,0,ATTN,0, kb_b, nullptr,0,0, Kb,0,ATTN, TOK,ATTN,din, 1.0f,0,1);
        g_launch(stream, hin,0,din, vb_w,0,ATTN,0, vb_b, nullptr,0,0, Vb,0,ATTN, TOK,ATTN,din, 1.0f,0,1);

        // forward attention: S = scale*Q@K^T; mask+softmax; ctx -> CTX[:, 0:64]
        g_launch(stream, Qf,tA,ATTN, Kf,tA,ATTN,1, nullptr, nullptr,0,0, S,tS,SEQ, SEQ,SEQ,ATTN, scl,0,BSZ);
        attn_softmax<<<dim3(SEQ,BSZ), 256, 0, stream>>>(S, whole_adj);
        g_launch(stream, S,tS,SEQ, Vf,tA,ATTN,0, nullptr, nullptr,0,0, CTX+0,tC,128, SEQ,ATTN,SEQ, 1.0f,0,BSZ);
        // backward attention -> CTX[:, 64:128]
        g_launch(stream, Qb,tA,ATTN, Kb,tA,ATTN,1, nullptr, nullptr,0,0, S,tS,SEQ, SEQ,SEQ,ATTN, scl,0,BSZ);
        attn_softmax<<<dim3(SEQ,BSZ), 256, 0, stream>>>(S, ancestor_adj);
        g_launch(stream, S,tS,SEQ, Vb,tA,ATTN,0, nullptr, nullptr,0,0, CTX+64,tC,128, SEQ,ATTN,SEQ, 1.0f,0,BSZ);

        // c = gelu(o(concat)); h = c + ff2(gelu(ff1(c)))
        g_launch(stream, CTX,0,128, o_w,0,HIDDEN,0, o_b, nullptr,0,0, Cb,0,HIDDEN, TOK,HIDDEN,128, 1.0f,1,1);
        g_launch(stream, Cb,0,HIDDEN, f1_w,0,FF,0, f1_b, nullptr,0,0, Fbuf,0,FF, TOK,FF,HIDDEN, 1.0f,1,1);
        g_launch(stream, Fbuf,0,FF, f2_w,0,HIDDEN,0, f2_b, Cb,0,HIDDEN, hout,0,HIDDEN, TOK,HIDDEN,FF, 1.0f,0,1);
    }

    // 3) LabelAttention2 (broadcast over labels collapses to kt_sum dot)
    g_launch(stream, structure,0,HIDDEN, laq_w,0,LA,0, laq_b, nullptr,0,0,
             Qt,0,LA, TOK,LA,HIDDEN, scl,0,1);
    kt_sum_kernel<<<1, 256, 0, stream>>>(label_emb, lak_w, lak_b, ktsum);
    la_scores<<<dim3(SEQ,BSZ), 128, 0, stream>>>(Qt, ktsum, adj, scores);
    softmax_row<<<dim3(BSZ), 256, 0, stream>>>(scores, SEQ);
    la_context<<<dim3(BSZ), 256, 0, stream>>>(scores, structure, features);

    // 4) head: gelu(feat(features)) -> out -> @ label_emb^T
    g_launch(stream, features,0,900, feat_w,0,HIDDEN,0, feat_b, nullptr,0,0,
             featH,0,HIDDEN, BSZ,HIDDEN,900, 1.0f,1,1);
    g_launch(stream, featH,0,HIDDEN, out_w,0,LBL,0, out_b, nullptr,0,0,
             outF,0,LBL, BSZ,LBL,HIDDEN, 1.0f,0,1);
    g_launch(stream, outF,0,LBL, label_emb,0,LBL,1, nullptr, nullptr,0,0,
             logits,0,NLAB, BSZ,NLAB,LBL, 1.0f,0,1);
}
